// HardNegativeContrastiveLoss_12051678232997
// MI455X (gfx1250) — matrix operand order repaired and checked by reading, not compiled, not run on hardware
//
#include <hip/hip_runtime.h>
#include <hip/hip_bf16.h>

typedef __attribute__((ext_vector_type(16))) _Float16 v16h;
typedef __attribute__((ext_vector_type(8)))  _Float16 v8h;
typedef __attribute__((ext_vector_type(4)))  _Float16 v4h;
typedef __attribute__((ext_vector_type(8)))  float    v8f;

#define BDIM 1024
#define QDIM 32
#define DDIM 512
#define NUM_HARD 512

// ---------------------------------------------------------------------------
// Kernel 1: f32 -> f16 conversion (vectorized, grid-stride)
// ---------------------------------------------------------------------------
__global__ __launch_bounds__(256) void cvt_f32_to_f16(const float* __restrict__ src,
                                                      _Float16* __restrict__ dst,
                                                      int n4) {
    int idx    = blockIdx.x * blockDim.x + threadIdx.x;
    int stride = gridDim.x * blockDim.x;
    const float4* s4 = reinterpret_cast<const float4*>(src);
    v4h* d4 = reinterpret_cast<v4h*>(dst);
    for (int i = idx; i < n4; i += stride) {
        float4 v = s4[i];
        v4h h;
        h[0] = (_Float16)v.x; h[1] = (_Float16)v.y;
        h[2] = (_Float16)v.z; h[3] = (_Float16)v.w;
        d4[i] = h;
    }
}

// ---------------------------------------------------------------------------
// Kernel 2: sim_max[i][j] = max_q fusion[i].target[j][q]  via WMMA f16
//   Block: 256 threads = 8 waves. Block covers 64 rows (4 i-tiles) x 8 j's.
//   Wave w handles j = jblk*8+w for all 32 q (two 16-wide N tiles), 4 i-tiles.
// ---------------------------------------------------------------------------
__global__ __launch_bounds__(256) void gemm_qmax(const _Float16* __restrict__ Af,   // [1024][512]
                                                 const _Float16* __restrict__ Tf,   // [1024][32][512]
                                                 float* __restrict__ simmax) {      // [1024][1024]
    const int lane = threadIdx.x & 31;
    const int wave = threadIdx.x >> 5;
    const int iblk = blockIdx.x;          // 16 blocks of 64 rows
    const int jblk = blockIdx.y;          // 128 blocks of 8 j's
    const int j    = jblk * 8 + wave;
    const int half = lane >> 4;           // 0 or 1
    const int l16  = lane & 15;

    // A fragment source rows: lane l16 -> matrix row l16 of each 16-row i-tile
    const _Float16* arow0 = Af + (iblk * 64 +  0 + l16) * DDIM;
    const _Float16* arow1 = Af + (iblk * 64 + 16 + l16) * DDIM;
    const _Float16* arow2 = Af + (iblk * 64 + 32 + l16) * DDIM;
    const _Float16* arow3 = Af + (iblk * 64 + 48 + l16) * DDIM;
    // B fragment source: lane l16 -> column q=l16 (tile0) and q=16+l16 (tile1)
    const _Float16* brow0 = Tf + (j * QDIM +      l16) * DDIM;
    const _Float16* brow1 = Tf + (j * QDIM + 16 + l16) * DDIM;

    v8f acc[4][2];
#pragma unroll
    for (int it = 0; it < 4; ++it)
#pragma unroll
        for (int n = 0; n < 2; ++n)
#pragma unroll
            for (int v = 0; v < 8; ++v) acc[it][n][v] = 0.0f;

    const int aoff0 = 16 * half;          // K = 16*half + {0..7}
    const int aoff1 = 16 * half + 8;      // K = 16*half+8 + {0..7}
    const int boff  = 16 * half;          // K = 16*half + {0..15}

    for (int k0 = 0; k0 < DDIM; k0 += 32) {
        // B fragments: 16 contiguous f16 (32B aligned)
        v16h bf0 = *reinterpret_cast<const v16h*>(brow0 + k0 + boff);
        v16h bf1 = *reinterpret_cast<const v16h*>(brow1 + k0 + boff);

#pragma unroll
        for (int it = 0; it < 4; ++it) {
            const _Float16* ar = (it == 0) ? arow0 : (it == 1) ? arow1
                               : (it == 2) ? arow2 : arow3;
            v8h a0 = *reinterpret_cast<const v8h*>(ar + k0 + aoff0);
            v8h a1 = *reinterpret_cast<const v8h*>(ar + k0 + aoff1);
            v16h af;
#pragma unroll
            for (int e = 0; e < 8; ++e) { af[e] = a0[e]; af[e + 8] = a1[e]; }

            acc[it][0] = __builtin_amdgcn_wmma_f32_16x16x32_f16(
                false, af, false, bf0, (short)0, acc[it][0], false, false);
            acc[it][1] = __builtin_amdgcn_wmma_f32_16x16x32_f16(
                false, af, false, bf1, (short)0, acc[it][1], false, false);
        }
    }

    // Max over q: combine the two N-tiles, then xor-reduce across 16 lanes.
    // C layout: VGPR v, lane L -> M = v + 8*(L/16), N = L%16.
#pragma unroll
    for (int it = 0; it < 4; ++it) {
        float m[8];
#pragma unroll
        for (int v = 0; v < 8; ++v) m[v] = fmaxf(acc[it][0][v], acc[it][1][v]);
#pragma unroll
        for (int mask = 1; mask <= 8; mask <<= 1)
#pragma unroll
            for (int v = 0; v < 8; ++v)
                m[v] = fmaxf(m[v], __shfl_xor(m[v], mask, 32));
        if (l16 == 0) {
            int rowbase = iblk * 64 + it * 16 + half * 8;
#pragma unroll
            for (int v = 0; v < 8; ++v)
                simmax[(rowbase + v) * BDIM + j] = m[v];
        }
    }
}

// ---------------------------------------------------------------------------
// Kernel 3: zero the output scalar
// ---------------------------------------------------------------------------
__global__ void zero_out(float* out) {
    if (threadIdx.x == 0 && blockIdx.x == 0) out[0] = 0.0f;
}

// ---------------------------------------------------------------------------
// Block reductions for the loss kernel
// ---------------------------------------------------------------------------
__device__ inline float blk_red_max(float v, float* red) {
    int tid = threadIdx.x;
    red[tid] = v; __syncthreads();
    for (int s = 128; s > 0; s >>= 1) {
        if (tid < s) red[tid] = fmaxf(red[tid], red[tid + s]);
        __syncthreads();
    }
    float r = red[0]; __syncthreads();
    return r;
}
__device__ inline float blk_red_sum(float v, float* red) {
    int tid = threadIdx.x;
    red[tid] = v; __syncthreads();
    for (int s = 128; s > 0; s >>= 1) {
        if (tid < s) red[tid] += red[tid + s];
        __syncthreads();
    }
    float r = red[0]; __syncthreads();
    return r;
}
__device__ inline int blk_red_sumi(int v, int* red) {
    int tid = threadIdx.x;
    red[tid] = v; __syncthreads();
    for (int s = 128; s > 0; s >>= 1) {
        if (tid < s) red[tid] += red[tid + s];
        __syncthreads();
    }
    int r = red[0]; __syncthreads();
    return r;
}

// ---------------------------------------------------------------------------
// Kernel 4: per-row losses.  One block (256 thr) per row i.
//   loss_std_i  = logsumexp(row) - row[i]
//   hard: exact top-512 of off-diagonal via threshold bisection, tie-corrected
//   loss_hard_i = log(exp(pos) + sum_{top512} exp) - pos
//   out += (loss_std_i + 0.5*loss_hard_i) / 1024
// ---------------------------------------------------------------------------
__global__ __launch_bounds__(256) void row_loss(const float* __restrict__ simmax,
                                                const float* __restrict__ temp_p,
                                                float* __restrict__ out) {
    __shared__ float sv[BDIM];
    __shared__ float redf[256];
    __shared__ int   redi[256];

    const int i   = blockIdx.x;
    const int tid = threadIdx.x;
    const float inv_t = 1.0f / temp_p[0];

    for (int jj = tid; jj < BDIM; jj += 256)
        sv[jj] = simmax[i * BDIM + jj] * inv_t;
    __syncthreads();

    const float pos = sv[i];

    // row statistics
    float mx_all = -3.0e38f, mx_off = -3.0e38f, mn_off = 3.0e38f;
    for (int jj = tid; jj < BDIM; jj += 256) {
        float v = sv[jj];
        mx_all = fmaxf(mx_all, v);
        if (jj != i) { mx_off = fmaxf(mx_off, v); mn_off = fminf(mn_off, v); }
    }
    mx_all = blk_red_max(mx_all, redf);
    mx_off = blk_red_max(mx_off, redf);
    mn_off = -blk_red_max(-mn_off, redf);

    // full-row logsumexp (includes diagonal)
    float se = 0.0f;
    for (int jj = tid; jj < BDIM; jj += 256) se += __expf(sv[jj] - mx_all);
    float sum_all  = blk_red_sum(se, redf);
    float lse_full = mx_all + __logf(sum_all);

    // bisection: threshold t s.t. count(off-diag >= t) == NUM_HARD
    float lo = mn_off, hi = mx_off;
    for (int iter = 0; iter < 40; ++iter) {
        float mid = 0.5f * (lo + hi);
        int c = 0;
        for (int jj = tid; jj < BDIM; jj += 256)
            if (jj != i && sv[jj] >= mid) c++;
        int ct = blk_red_sumi(c, redi);
        if (ct >= NUM_HARD) lo = mid; else hi = mid;
    }
    const float t  = lo;
    const float mm = fmaxf(pos, mx_off);

    int c = 0; float s = 0.0f;
    for (int jj = tid; jj < BDIM; jj += 256) {
        if (jj != i && sv[jj] >= t) { c++; s += __expf(sv[jj] - mm); }
    }
    int   ctot = blk_red_sumi(c, redi);
    float stot = blk_red_sum(s, redf);
    stot -= (float)(ctot - NUM_HARD) * __expf(t - mm);   // tie correction

    if (tid == 0) {
        float hard_lse = mm + __logf(__expf(pos - mm) + stot);
        float li = (lse_full - pos) + 0.5f * (hard_lse - pos);
        atomicAdd(out, li * (1.0f / (float)BDIM));
    }
}

// ---------------------------------------------------------------------------
extern "C" void kernel_launch(void* const* d_in, const int* in_sizes, int n_in,
                              void* d_out, int out_size, void* d_ws, size_t ws_size,
                              hipStream_t stream) {
    const float* fusion = (const float*)d_in[0];   // [1024][512]
    const float* target = (const float*)d_in[1];   // [1024][32][512]
    const float* temp   = (const float*)d_in[2];   // scalar
    float* out = (float*)d_out;

    // workspace layout
    char* ws = (char*)d_ws;
    _Float16* Af = (_Float16*)ws;                               // 1 MiB
    _Float16* Tf = (_Float16*)(ws + (1u << 20));                // 32 MiB
    float* simmax = (float*)(ws + (1u << 20) + (32u << 20));    // 4 MiB

    // 1) convert both inputs to f16
    {
        int n4f = BDIM * DDIM / 4;          // 131072
        cvt_f32_to_f16<<<(n4f + 255) / 256, 256, 0, stream>>>(fusion, Af, n4f);
        int n4t = BDIM * QDIM * DDIM / 4;   // 4194304
        cvt_f32_to_f16<<<4096, 256, 0, stream>>>(target, Tf, n4t);
    }

    // 2) WMMA GEMM with q-max fusion
    gemm_qmax<<<dim3(16, 128), 256, 0, stream>>>(Af, Tf, simmax);

    // 3) scalar reduction
    zero_out<<<1, 32, 0, stream>>>(out);
    row_loss<<<BDIM, 256, 0, stream>>>(simmax, temp, out);
}
